// LowRankSSM_26843545600398
// MI455X (gfx1250) — matrix operand order repaired and checked by reading, not compiled, not run on hardware
//
#include <hip/hip_runtime.h>

// ---------------------------------------------------------------------------
// Types for CDNA5 WMMA (wave32): D(f32 16x16) = A(bf16 16x32) * B(bf16 32x16) + C
// ---------------------------------------------------------------------------
typedef __attribute__((ext_vector_type(16))) __bf16 v16bf;
typedef __attribute__((ext_vector_type(8)))  __bf16 v8bf;
typedef __attribute__((ext_vector_type(8)))  float  v8f;
typedef __attribute__((ext_vector_type(4)))  float  v4f;

#define WMMA_BF16(a, b, c) \
  __builtin_amdgcn_wmma_f32_16x16x32_bf16(false, (a), false, (b), (short)0, (c), false, false)

// Native conversion: gfx1250 has hardware bf16 support, let clang lower it.
static __device__ __forceinline__ __bf16 f2bf(float f) { return (__bf16)f; }

// ---------------------------------------------------------------------------
// Problem constants
// ---------------------------------------------------------------------------
#define BB   8
#define TT   4096
#define PP   1024
#define PR   128
#define MM   (BB * TT)        // 32768 rows
#define CHK  256              // scan chunk length
#define NC   (TT / CHK)       // 16 chunks per sequence

// ---------------------------------------------------------------------------
// Kernel 0: pack an f32 [K x N] weight into bf16 WMMA B-fragment order.
// Fragment block (tk, tn) = 32 lanes x 16 bf16, contiguous 32B per lane.
// Lane l: N = tn*16 + (l&15); element e: K = tk*32 + (e<8 ? 8*(l>>4)+e : 16+8*(l>>4)+e-8).
// ---------------------------------------------------------------------------
__global__ void pack_frag_kernel(const float* __restrict__ src,
                                 __bf16* __restrict__ dst, int K, int N) {
  int o = blockIdx.x * blockDim.x + threadIdx.x;
  if (o >= K * N) return;
  int e      = o & 15;
  int lane   = (o >> 4) & 31;
  int ntiles = N >> 4;
  int tn     = (o >> 9) % ntiles;
  int tk     = o / (512 * ntiles);
  int col    = tn * 16 + (lane & 15);
  int k      = tk * 32 + ((e < 8) ? 8 * (lane >> 4) + e : 16 + 8 * (lane >> 4) + (e - 8));
  dst[o] = f2bf(src[(size_t)k * N + col]);
}

// ---------------------------------------------------------------------------
// Kernel 1: fused  U = t @ V_r ; b_all = (f*t) @ V_b ; a_all = clip(sig(U@Wl+bl))
// Block = 256 threads (8 waves). Wave w owns rows [blk*128 + w*16, +16).
// t / f_gate are streamed non-temporally (read-once, 268 MB total) so the
// L2-resident weight fragments and Abuf/Bbuf stay hot.
// ---------------------------------------------------------------------------
__global__ __launch_bounds__(256)
void fused_ab_kernel(const float* __restrict__ t, const float* __restrict__ f,
                     const __bf16* __restrict__ VrF, const __bf16* __restrict__ VbF,
                     const __bf16* __restrict__ WlF, const float* __restrict__ bl,
                     float* __restrict__ Abuf, float* __restrict__ Bbuf) {
  const int wave = threadIdx.x >> 5;
  const int lane = threadIdx.x & 31;
  const int lh   = lane & 15;
  const int hi   = lane >> 4;               // lane group: 0 or 1
  const int b0   = hi ? 8 : 0;              // A-fragment K base for this lane group
  const int rowBase = blockIdx.x * 128 + wave * 16;
  const int r       = rowBase + lh;

  __shared__ __bf16 ldsU[8][16][PR];        // 32 KB: per-wave U tile (row-major)

  v8f accU[8] = {}, accB[8] = {};

  const float* trow = t + (size_t)r * PP;
  const float* frow = f + (size_t)r * PP;

  for (int kk = 0; kk < PP; kk += 32) {
    // ---- A fragments: t and f*t (bf16 16x32, lane-striped) ----
    v4f t0 = __builtin_nontemporal_load((const v4f*)(trow + kk + b0));
    v4f t1 = __builtin_nontemporal_load((const v4f*)(trow + kk + b0 + 4));
    v4f t2 = __builtin_nontemporal_load((const v4f*)(trow + kk + b0 + 16));
    v4f t3 = __builtin_nontemporal_load((const v4f*)(trow + kk + b0 + 20));
    v4f f0 = __builtin_nontemporal_load((const v4f*)(frow + kk + b0));
    v4f f1 = __builtin_nontemporal_load((const v4f*)(frow + kk + b0 + 4));
    v4f f2 = __builtin_nontemporal_load((const v4f*)(frow + kk + b0 + 16));
    v4f f3 = __builtin_nontemporal_load((const v4f*)(frow + kk + b0 + 20));

    v16bf at, aft;
#pragma unroll
    for (int e = 0; e < 4; ++e) {
      at[e]      = f2bf(t0[e]);  aft[e]      = f2bf(t0[e] * f0[e]);
      at[4 + e]  = f2bf(t1[e]);  aft[4 + e]  = f2bf(t1[e] * f1[e]);
      at[8 + e]  = f2bf(t2[e]);  aft[8 + e]  = f2bf(t2[e] * f2[e]);
      at[12 + e] = f2bf(t3[e]);  aft[12 + e] = f2bf(t3[e] * f3[e]);
    }

    if (kk + 32 < PP) {
      __builtin_prefetch(trow + kk + 32, 0, 0);
      __builtin_prefetch(frow + kk + 32, 0, 0);
    }

    const int tk = kk >> 5;
#pragma unroll
    for (int j = 0; j < 8; ++j) {           // 8 N-tiles of 16 (P_R = 128)
      v16bf br = *(const v16bf*)(VrF + ((size_t)(tk * 8 + j) * 512) + lane * 16);
      accU[j] = WMMA_BF16(at, br, accU[j]);
      v16bf bb = *(const v16bf*)(VbF + ((size_t)(tk * 8 + j) * 512) + lane * 16);
      accB[j] = WMMA_BF16(aft, bb, accB[j]);
    }
  }

  // ---- store b_all (C-layout: element g -> row g + 8*hi, col = j*16 + lh) ----
#pragma unroll
  for (int j = 0; j < 8; ++j)
#pragma unroll
    for (int g = 0; g < 8; ++g)
      Bbuf[(size_t)(rowBase + g + hi * 8) * PR + j * 16 + lh] = accB[j][g];

  // ---- re-fragment U via LDS for the small GEMM U @ W_lambda ----
#pragma unroll
  for (int j = 0; j < 8; ++j)
#pragma unroll
    for (int g = 0; g < 8; ++g)
      ldsU[wave][g + hi * 8][j * 16 + lh] = f2bf(accU[j][g]);
  __syncthreads();

  v8f accA[8] = {};
#pragma unroll
  for (int kk = 0; kk < PR; kk += 32) {
    const __bf16* up = &ldsU[wave][lh][kk + b0];
    v8bf lo = *(const v8bf*)up;
    v8bf hv = *(const v8bf*)(up + 16);
    v16bf au;
#pragma unroll
    for (int e = 0; e < 8; ++e) { au[e] = lo[e]; au[8 + e] = hv[e]; }
    const int tk = kk >> 5;
#pragma unroll
    for (int j = 0; j < 8; ++j) {
      v16bf bw = *(const v16bf*)(WlF + ((size_t)(tk * 8 + j) * 512) + lane * 16);
      accA[j] = WMMA_BF16(au, bw, accA[j]);
    }
  }

  // ---- a = clip(sigmoid(acc + b_lambda), 0, 1) ----
#pragma unroll
  for (int j = 0; j < 8; ++j) {
    float blv = bl[j * 16 + lh];
#pragma unroll
    for (int g = 0; g < 8; ++g) {
      float x = accA[j][g] + blv;
      float a = 1.0f / (1.0f + __expf(-x));
      a = fminf(fmaxf(a, 0.0f), 1.0f);
      Abuf[(size_t)(rowBase + g + hi * 8) * PR + j * 16 + lh] = a;
    }
  }
}

// ---------------------------------------------------------------------------
// Chunked parallel scan:  s_t = a_t * s_{t-1} + b_t   over T, per (b, r) lane.
// ---------------------------------------------------------------------------
__global__ void scan_pass1(const float* __restrict__ A, const float* __restrict__ Bv,
                           float* __restrict__ cA, float* __restrict__ cS) {
  int idx = blockIdx.x * blockDim.x + threadIdx.x;   // B*NC*PR = 16384
  int rr = idx & (PR - 1);
  int c  = (idx >> 7) & (NC - 1);
  int b  = idx >> 11;
  size_t base = ((size_t)b * TT + (size_t)c * CHK) * PR + rr;
  float pA = 1.0f, s = 0.0f;
  for (int tt = 0; tt < CHK; ++tt) {
    float a  = A[base + (size_t)tt * PR];
    float bb = Bv[base + (size_t)tt * PR];
    pA *= a;
    s = a * s + bb;
  }
  cA[idx] = pA;
  cS[idx] = s;
}

__global__ void scan_pass2(const float* __restrict__ cA, const float* __restrict__ cS,
                           float* __restrict__ carryIn, float* __restrict__ sfinal) {
  int idx = blockIdx.x * blockDim.x + threadIdx.x;   // B*PR = 1024
  int rr = idx & (PR - 1);
  int b  = idx >> 7;
  float cin = 0.0f;
  for (int c = 0; c < NC; ++c) {
    int i = (b * NC + c) * PR + rr;
    carryIn[i] = cin;
    cin = cA[i] * cin + cS[i];
  }
  sfinal[b * PR + rr] = cin;                         // s_final -> output tail
}

__global__ void scan_pass3(const float* __restrict__ A, const float* __restrict__ Bv,
                           const float* __restrict__ carryIn, __bf16* __restrict__ Sout) {
  int idx = blockIdx.x * blockDim.x + threadIdx.x;   // 16384
  int rr = idx & (PR - 1);
  int c  = (idx >> 7) & (NC - 1);
  int b  = idx >> 11;
  size_t base = ((size_t)b * TT + (size_t)c * CHK) * PR + rr;
  float s = carryIn[idx];
  for (int tt = 0; tt < CHK; ++tt) {
    float a  = A[base + (size_t)tt * PR];
    float bb = Bv[base + (size_t)tt * PR];
    s = a * s + bb;
    Sout[base + (size_t)tt * PR] = f2bf(s);
  }
}

// ---------------------------------------------------------------------------
// Kernel 4: tilde_t = s_seq(bf16) @ V_o(bf16-frag), f32 out. 128x128 per block.
// Output is written non-temporally (write-once, 134 MB).
// ---------------------------------------------------------------------------
__global__ __launch_bounds__(256)
void out_gemm_kernel(const __bf16* __restrict__ S, const __bf16* __restrict__ VoF,
                     float* __restrict__ out) {
  const int wave = threadIdx.x >> 5;
  const int lane = threadIdx.x & 31;
  const int lh   = lane & 15;
  const int hi   = lane >> 4;
  const int b0   = hi ? 8 : 0;
  const int rowBase = blockIdx.x * 128 + wave * 16;
  const int colBase = blockIdx.y * 128;
  const int r       = rowBase + lh;

  const __bf16* srow = S + (size_t)r * PR;
  v8f acc[8] = {};

#pragma unroll
  for (int kk = 0; kk < PR; kk += 32) {
    v8bf lo = *(const v8bf*)(srow + kk + b0);
    v8bf hv = *(const v8bf*)(srow + kk + b0 + 16);
    v16bf a;
#pragma unroll
    for (int e = 0; e < 8; ++e) { a[e] = lo[e]; a[8 + e] = hv[e]; }
    const int tk = kk >> 5;
#pragma unroll
    for (int j = 0; j < 8; ++j) {
      int tn = (colBase >> 4) + j;                   // of 64 N-tiles (P = 1024)
      v16bf bv = *(const v16bf*)(VoF + ((size_t)(tk * 64 + tn) * 512) + lane * 16);
      acc[j] = WMMA_BF16(a, bv, acc[j]);
    }
  }

#pragma unroll
  for (int j = 0; j < 8; ++j)
#pragma unroll
    for (int g = 0; g < 8; ++g)
      __builtin_nontemporal_store(
          acc[j][g],
          out + (size_t)(rowBase + g + hi * 8) * PP + colBase + j * 16 + lh);
}

// ---------------------------------------------------------------------------
// Host launcher
// ---------------------------------------------------------------------------
extern "C" void kernel_launch(void* const* d_in, const int* in_sizes, int n_in,
                              void* d_out, int out_size, void* d_ws, size_t ws_size,
                              hipStream_t stream) {
  const float* t  = (const float*)d_in[0];   // [B,T,P]
  const float* fg = (const float*)d_in[1];   // [B,T,P]
  const float* Vr = (const float*)d_in[2];   // [P,P_R]
  const float* Wl = (const float*)d_in[3];   // [P_R,P_R]
  const float* bl = (const float*)d_in[4];   // [P_R]
  const float* Vb = (const float*)d_in[5];   // [P,P_R]
  const float* Vo = (const float*)d_in[6];   // [P_R,P]
  float* out = (float*)d_out;                // tilde_t [B,T,P] ++ s_final [B,P_R]

  char* ws = (char*)d_ws;
  size_t off = 0;
  auto alloc = [&](size_t bytes) -> void* {
    void* p = ws + off;
    off += (bytes + 255) & ~(size_t)255;
    return p;
  };
  __bf16* VrF  = (__bf16*)alloc((size_t)PP * PR * 2);
  __bf16* VbF  = (__bf16*)alloc((size_t)PP * PR * 2);
  __bf16* WlF  = (__bf16*)alloc((size_t)PR * PR * 2);
  __bf16* VoF  = (__bf16*)alloc((size_t)PR * PP * 2);
  float*  Abuf = (float*)alloc((size_t)MM * PR * 4);
  float*  Bbuf = (float*)alloc((size_t)MM * PR * 4);
  __bf16* Sseq = (__bf16*)alloc((size_t)MM * PR * 2);
  float*  cA   = (float*)alloc((size_t)BB * NC * PR * 4);
  float*  cS   = (float*)alloc((size_t)BB * NC * PR * 4);
  float*  cIn  = (float*)alloc((size_t)BB * NC * PR * 4);

  pack_frag_kernel<<<(PP * PR + 255) / 256, 256, 0, stream>>>(Vr, VrF, PP, PR);
  pack_frag_kernel<<<(PP * PR + 255) / 256, 256, 0, stream>>>(Vb, VbF, PP, PR);
  pack_frag_kernel<<<(PR * PR + 255) / 256, 256, 0, stream>>>(Wl, WlF, PR, PR);
  pack_frag_kernel<<<(PR * PP + 255) / 256, 256, 0, stream>>>(Vo, VoF, PR, PP);

  fused_ab_kernel<<<MM / 128, 256, 0, stream>>>(t, fg, VrF, VbF, WlF, bl, Abuf, Bbuf);

  scan_pass1<<<(BB * NC * PR) / 256, 256, 0, stream>>>(Abuf, Bbuf, cA, cS);
  scan_pass2<<<(BB * PR) / 256, 256, 0, stream>>>(cA, cS, cIn,
                                                  out + (size_t)BB * TT * PP);
  scan_pass3<<<(BB * NC * PR) / 256, 256, 0, stream>>>(Abuf, Bbuf, cIn, Sseq);

  dim3 g4(MM / 128, PP / 128);
  out_gemm_kernel<<<g4, 256, 0, stream>>>(Sseq, VoF, out);
}
